// Dataset1V7Table5Redo_69741678952822
// MI455X (gfx1250) — matrix operand order repaired and checked by reading, compiled, not run on hardware
//
#include <hip/hip_runtime.h>
#include <stdint.h>

typedef _Float16 half_t;
typedef __attribute__((ext_vector_type(4)))  _Float16 v4h;
typedef __attribute__((ext_vector_type(8)))  _Float16 v8h;
typedef __attribute__((ext_vector_type(16))) _Float16 v16h;
typedef __attribute__((ext_vector_type(8)))  float    v8f;
typedef __attribute__((ext_vector_type(4)))  unsigned int v4u;
typedef __attribute__((ext_vector_type(8)))  int      v8i;
typedef __attribute__((ext_vector_type(4)))  int      v4i;

namespace {
constexpr int TILE_M = 128;   // rows per workgroup
constexpr int DD     = 256;   // input dim
constexpr int HH     = 128;   // hidden dim
constexpr int NT     = 256;   // threads per workgroup (8 waves of 32)
// LDS carve-up (bytes)
constexpr size_t XBUF_BYTES   = (size_t)TILE_M * DD * sizeof(float); // 128KB f32 x-tile (TDM dest),
                                                                     // reused as T/T2 f16 [128][128]
constexpr size_t SW1_BYTE_OFF = XBUF_BYTES;                          // 64KB f16 W1 (native o,d)
constexpr size_t SW2_BYTE_OFF = XBUF_BYTES + (size_t)HH * DD * 2;    // 32KB f16 W2 (native o,k)
constexpr size_t SMEM_BYTES   = SW2_BYTE_OFF + (size_t)HH * HH * 2;  // 224KB total
}

__device__ __forceinline__ v16h make16(v8h lo, v8h hi) {
  return __builtin_shufflevector(lo, hi, 0,1,2,3,4,5,6,7,8,9,10,11,12,13,14,15);
}

// A-matrix 16x32 f16 fragment from an f32 LDS tile (convert at load time).
// Layout (ISA 7.12.2): lane holds row (lane&15); lanes 0-15: K={0..7,16..23},
// lanes 16-31: K={8..15,24..31}.
__device__ __forceinline__ v16h frag_A_f32(const float* s, int row0, int k0, int ld, int lane) {
  const int r  = lane & 15;
  const int ka = (lane >> 4) << 3;       // 0 or 8
  const float* p = s + (row0 + r) * ld + k0 + ka;
  v16h a;
  #pragma unroll
  for (int i = 0; i < 8; ++i) {
    a[i]     = (half_t)p[i];
    a[8 + i] = (half_t)p[16 + i];
  }
  return a;
}

// A-matrix 16x32 f16 fragment from an f16 LDS tile.
__device__ __forceinline__ v16h frag_A(const half_t* s, int row0, int k0, int ld, int lane) {
  const int r  = lane & 15;
  const int ka = (lane >> 4) << 3;       // 0 or 8
  const half_t* p = s + (row0 + r) * ld + k0 + ka;
  return make16(*(const v8h*)(p), *(const v8h*)(p + 16));
}

// B-matrix 32x16 f16 fragment with B[k][n] = W[n][k] (weights kept in native (o,k) order):
// lane holds column n = col0 + (lane&15); lanes 0-15: K={0..7,16..23}, lanes 16-31: K={8..15,24..31}.
__device__ __forceinline__ v16h frag_B(const half_t* s, int col0, int k0, int ld, int lane) {
  const int c  = lane & 15;
  const int kb = (lane >> 4) << 3;       // 0 or 8
  const half_t* p = s + (col0 + c) * ld + k0 + kb;
  return make16(*(const v8h*)(p), *(const v8h*)(p + 16));
}

__global__ __launch_bounds__(NT, 1)
void fused_setvalued_mlp_kernel(const float* __restrict__ x,     // (65536, 256)
                                const float* __restrict__ W1,    // (128, 256)
                                const float* __restrict__ W2,    // (128, 128)
                                const float* __restrict__ Wout,  // (128)
                                const float* __restrict__ soa,   // (32)
                                const float* __restrict__ sob,   // (32)
                                float* __restrict__ out)         // (65536)
{
  extern __shared__ char smem[];
  float*  xF  = (float*)smem;                       // [128][256] f32 (TDM destination)
  half_t* sT  = (half_t*)smem;                      // [128][128] f16 overlay (post-GEMM1)
  half_t* sW1 = (half_t*)(smem + SW1_BYTE_OFF);     // [128][256]
  half_t* sW2 = (half_t*)(smem + SW2_BYTE_OFF);     // [128][128]

  const int tid  = threadIdx.x;
  const int lane = tid & 31;
  const int wv   = tid >> 5;
  const int row0 = wv << 4;

  // ---------- TDM: async tensor DMA of the x tile (f32 256x128) into LDS ----------
  // Issued once (wave 0); overlapped with the VALU f32->f16 staging of W1/W2 below.
  if (wv == 0) {
    const unsigned ldsa = __builtin_amdgcn_groupstaticsize();      // dynamic-LDS base offset
    const unsigned long long ga =
        (unsigned long long)(uintptr_t)(x + (size_t)blockIdx.x * TILE_M * DD);
    // D# group 0: count=1 | lds_addr | global_addr[56:0] | type=2 ("image")
    v4u g0 = { 1u,
               ldsa,
               (unsigned)(ga & 0xFFFFFFFFull),
               (unsigned)((ga >> 32) & 0x1FFFFFFull) | (2u << 30) };
    // D# group 1: data_size=4B; tensor_dim0=256; tensor_dim1=128;
    //             tile_dim0=256; tile_dim1=128; tensor_dim0_stride=256.
    v8i g1 = { (int)(2u << 16),                       // [17:16] data_size = 2 (4 bytes)
               (int)((unsigned)DD << 16),             // [31:16] tensor_dim0 lo16 = 256
               (int)((unsigned)TILE_M << 16),         // [15:0] dim0 hi = 0 | [31:16] tensor_dim1 lo16
               (int)((unsigned)DD << 16),             // [15:0] dim1 hi = 0 | [31:16] tile_dim0 = 256
               (int)TILE_M,                           // [15:0] tile_dim1 = 128 | tile_dim2 = 0
               (int)DD,                               // tensor_dim0_stride[31:0] = 256
               0, 0 };                                // stride0 hi | tensor_dim1_stride = 0 (2D)
    v4i gz = { 0, 0, 0, 0 };                          // groups 2/3 unused for 2D tile
#if defined(__clang_major__) && (__clang_major__ >= 23)
    v8i gz8 = { 0, 0, 0, 0, 0, 0, 0, 0 };
    __builtin_amdgcn_tensor_load_to_lds(g0, g1, gz, gz, gz8, 0);
#else
    __builtin_amdgcn_tensor_load_to_lds(g0, g1, gz, gz, 0);
#endif
  }

  // ---------- cooperative f32 -> f16 staging of W1 / W2 (L2-resident) ----------
  {
    const float4* w1g = (const float4*)W1;
    for (int i = tid; i < HH * (DD / 4); i += NT) {
      float4 v = w1g[i];
      v4h h = { (half_t)v.x, (half_t)v.y, (half_t)v.z, (half_t)v.w };
      *(v4h*)(sW1 + (size_t)i * 4) = h;
    }
    const float4* w2g = (const float4*)W2;
    for (int i = tid; i < HH * (HH / 4); i += NT) {
      float4 v = w2g[i];
      v4h h = { (half_t)v.x, (half_t)v.y, (half_t)v.z, (half_t)v.w };
      *(v4h*)(sW2 + (size_t)i * 4) = h;
    }
  }

  if (wv == 0) __builtin_amdgcn_s_wait_tensorcnt((short)0);  // TDM done before barrier
  __syncthreads();

  // ---------- GEMM1: T = tanh(X * W1^T), 128x256 . 256x128 (A converted f32->f16) ----------
  v8f acc[8] = {};
  for (int k0 = 0; k0 < DD; k0 += 32) {
    v16h a = frag_A_f32(xF, row0, k0, DD, lane);
    #pragma unroll
    for (int t = 0; t < 8; ++t) {
      v16h b = frag_B(sW1, t * 16, k0, DD, lane);
      acc[t] = __builtin_amdgcn_wmma_f32_16x16x32_f16(false, a, false, b,
                                                      (short)0, acc[t], false, false);
    }
  }
  __syncthreads();  // all waves done reading xF before T overlays it

  // C/D layout: VGPR r -> row (r + 8*(lane>=16)), col = lane&15 within the 16x16 tile.
  {
    const int c   = lane & 15;
    const int hi8 = (lane >> 4) << 3;
    #pragma unroll
    for (int t = 0; t < 8; ++t)
      #pragma unroll
      for (int r = 0; r < 8; ++r)
        sT[(row0 + hi8 + r) * HH + t * 16 + c] = (half_t)tanhf(acc[t][r]);
  }
  __syncthreads();

  // ---------- GEMM2: T2 = tanh(T * W2^T), 128x128 . 128x128 ----------
  v8f acc2[8] = {};
  for (int k0 = 0; k0 < HH; k0 += 32) {
    v16h a = frag_A(sT, row0, k0, HH, lane);
    #pragma unroll
    for (int t = 0; t < 8; ++t) {
      v16h b = frag_B(sW2, t * 16, k0, HH, lane);
      acc2[t] = __builtin_amdgcn_wmma_f32_16x16x32_f16(false, a, false, b,
                                                       (short)0, acc2[t], false, false);
    }
  }
  __syncthreads();  // all waves done reading T

  {
    const int c   = lane & 15;
    const int hi8 = (lane >> 4) << 3;
    #pragma unroll
    for (int t = 0; t < 8; ++t)
      #pragma unroll
      for (int r = 0; r < 8; ++r)
        sT[(row0 + hi8 + r) * HH + t * 16 + c] = (half_t)tanhf(acc2[t][r]);
  }
  __syncthreads();

  // ---------- epilogue: closed-form path collapse ----------
  // a = Wout . T2_row ; s = sum_h sob[h]*sin(2*pi*soa[h]*a/7)
  // out = sign(s) * 4a / (4 + 5*exp(-|s|)) ; 0 if s == 0.
  if (tid < TILE_M) {
    float av = 0.f;
    #pragma unroll 8
    for (int o = 0; o < HH; ++o)
      av += (float)sT[tid * HH + o] * Wout[o];

    const float kph = 6.28318530717958647692f / 7.0f;
    float s = 0.f;
    #pragma unroll 8
    for (int h = 0; h < 32; ++h)
      s += sob[h] * __sinf(kph * soa[h] * av);

    float res = 0.f;
    if (s != 0.f) {
      float e = __expf(-fabsf(s));
      float v = 4.f * av / (4.f + 5.f * e);
      res = (s > 0.f) ? v : -v;
    }
    out[(size_t)blockIdx.x * TILE_M + tid] = res;
  }
}

extern "C" void kernel_launch(void* const* d_in, const int* in_sizes, int n_in,
                              void* d_out, int out_size, void* d_ws, size_t ws_size,
                              hipStream_t stream) {
  (void)n_in; (void)out_size; (void)d_ws; (void)ws_size;
  const float* x    = (const float*)d_in[0];
  const float* W1   = (const float*)d_in[1];
  const float* W2   = (const float*)d_in[2];
  const float* Wout = (const float*)d_in[3];
  // d_in[4..7] = s1a, s1b, s2a, s2b are dead code (their prune never triggers).
  const float* soa  = (const float*)d_in[8];
  const float* sob  = (const float*)d_in[9];
  float* out = (float*)d_out;

  const int rows   = in_sizes[0] / DD;     // 65536
  const int blocks = rows / TILE_M;        // 512

  (void)hipFuncSetAttribute(reinterpret_cast<const void*>(&fused_setvalued_mlp_kernel),
                            hipFuncAttributeMaxDynamicSharedMemorySize, (int)SMEM_BYTES);
  fused_setvalued_mlp_kernel<<<blocks, NT, SMEM_BYTES, stream>>>(
      x, W1, W2, Wout, soa, sob, out);
}
